// MSDeformAttn_49177375539411
// MI455X (gfx1250) — matrix operand order repaired and checked by reading, not compiled, not run on hardware
//
#include <hip/hip_runtime.h>
#include <hip/hip_bf16.h>

// MS-Deformable Attention for gfx1250 (MI455X), wave32 + WMMA.
//
// Pipeline (all on `stream`, serialized by stream order):
//   1. pack_a(input_flatten) -> A1 (f16 WMMA A-fragment layout)
//   2. pack_b(value_proj_w)  -> B1 (f16 WMMA B-fragment layout)
//   3. wmma_gemm -> value[M,256]            (v_wmma_f32_16x16x32_f16)
//   4. pack_b(samp_off_w); 5. pack_a(query); 6. wmma_gemm -> offs[M,256]
//   7. pack_b(attn_w_w);                     8. wmma_gemm -> attn[M,128]
//   9. softmax over 16 (levels*points) per (q,head)
//  10. bilinear sampling: one wave per (q,head), lane = head-dim channel
//  11. pack_a(tmp); 12. pack_b(out_proj_w); 13. wmma_gemm -> d_out[M,256]

typedef __attribute__((ext_vector_type(16))) _Float16 v16h;
typedef __attribute__((ext_vector_type(8)))  float    v8f;

#define D_MODEL   256
#define N_HEADS   8
#define N_LEVELS  4
#define N_POINTS  4
#define HEAD_DIM  32

// ---------------------------------------------------------------------------
// Pack f32 activations [M,256] into the CDNA5 16-bit A-matrix 16x32 fragment
// layout (ISA 7.12.2): per lane, row = lane%16,
//   lanes 0-15:  halfs 0..7 -> K 0..7,  halfs 8..15 -> K 16..23
//   lanes 16-31: halfs 0..7 -> K 8..15, halfs 8..15 -> K 24..31
// Flat layout: [mtile][kstep(8)][lane(32)][16 halfs]
__global__ void msda_pack_a_kernel(const float* __restrict__ A,
                                   _Float16* __restrict__ out,
                                   int M, long total) {
  long t = (long)blockIdx.x * blockDim.x + threadIdx.x;
  if (t >= total) return;
  int i    = (int)(t & 15);
  int lane = (int)((t >> 4) & 31);
  int s    = (int)((t >> 9) & 7);
  long tile = t >> 12;
  int row = (int)tile * 16 + (lane & 15);
  int kl  = ((i < 8) ? i : (8 + i)) + ((lane >= 16) ? 8 : 0);
  int k   = s * 32 + kl;
  float v = (row < M) ? A[(long)row * D_MODEL + k] : 0.0f;
  out[t] = (_Float16)v;
}

// Pack f32 weights [256,N] (row-major, K x N) into the 16-bit B-matrix 32x16
// fragment layout: per lane, col = lane%16,
//   lanes 0-15: halfs 0..7 -> K 0..7, 8..15 -> K 16..23 ; lanes 16-31: halfs 0..7 -> K 8..15, 8..15 -> K 24..31
// Flat layout: [ntile][kstep(8)][lane(32)][16 halfs]
__global__ void msda_pack_b_kernel(const float* __restrict__ W,
                                   _Float16* __restrict__ out,
                                   int N, long total) {
  long t = (long)blockIdx.x * blockDim.x + threadIdx.x;
  if (t >= total) return;
  int i    = (int)(t & 15);
  int lane = (int)((t >> 4) & 31);
  int s    = (int)((t >> 9) & 7);
  long tile = t >> 12;
  int col = (int)tile * 16 + (lane & 15);
  int kl  = ((i < 8) ? i : (8 + i)) + ((lane >= 16) ? 8 : 0);
  int k   = s * 32 + kl;
  out[t] = (_Float16)W[(long)k * N + col];
}

// C[M,N] = A[M,256] @ W[256,N] + bias, f16 WMMA, f32 accumulate.
// Block = 256 threads = 8 waves; wave w owns C tile (blockIdx.x, blockIdx.y*8+w).
// Full EXEC throughout the WMMA loop (wave32, no divergence before stores).
__global__ void msda_wmma_gemm_kernel(const v16h* __restrict__ A,
                                      const v16h* __restrict__ B,
                                      const float* __restrict__ bias,
                                      float* __restrict__ C,
                                      int M, int N) {
  int lane  = threadIdx.x & 31;
  int wave  = threadIdx.x >> 5;
  int mtile = blockIdx.x;
  int ntile = blockIdx.y * 8 + wave;

  long abase = ((long)mtile * 8) * 32 + lane;  // v16h units
  long bbase = ((long)ntile * 8) * 32 + lane;

  v8f c = {};
#pragma unroll
  for (int s = 0; s < 8; ++s) {        // K = 8 * 32 = 256
    v16h a = A[abase + s * 32];
    v16h b = B[bbase + s * 32];
    c = __builtin_amdgcn_wmma_f32_16x16x32_f16(
        /*neg_a=*/false, a, /*neg_b=*/false, b,
        /*c_mod=*/(short)0, c, /*reuse_a=*/false, /*reuse_b=*/false);
  }

  // C/D layout: VGPR r, lanes 0-15 -> (M=r, N=lane); lanes 16-31 -> (M=r+8).
  int col   = ntile * 16 + (lane & 15);
  float bv  = bias[col];
  int rbase = mtile * 16 + ((lane >= 16) ? 8 : 0);
#pragma unroll
  for (int r = 0; r < 8; ++r) {
    int row = rbase + r;
    if (row < M) C[(long)row * N + col] = c[r] + bv;
  }
}

// Softmax over the 16 (level,point) logits per (q, head). One thread per row.
__global__ void msda_softmax16_kernel(float* __restrict__ attn, int M) {
  long t = (long)blockIdx.x * blockDim.x + threadIdx.x;
  if (t >= (long)M * N_HEADS) return;
  float* p = attn + t * 16;
  float m = p[0];
#pragma unroll
  for (int i = 1; i < 16; ++i) m = fmaxf(m, p[i]);
  float e[16];
  float s = 0.0f;
#pragma unroll
  for (int i = 0; i < 16; ++i) { e[i] = __expf(p[i] - m); s += e[i]; }
  float inv = 1.0f / s;
#pragma unroll
  for (int i = 0; i < 16; ++i) p[i] = e[i] * inv;
}

// Deformable bilinear sampling + attention-weighted accumulation.
// One wave per (q, head); lane = channel within HEAD_DIM(=32), so every corner
// gather is one coalesced 128B load. value[20MB] is L2-resident (192MB L2).
__global__ void msda_sample_kernel(const float* __restrict__ value,
                                   const float* __restrict__ offs,
                                   const float* __restrict__ attn,
                                   const float* __restrict__ refp,
                                   const int* __restrict__ shapes,
                                   const int* __restrict__ starts,
                                   float* __restrict__ tmp, int M) {
  int lane = threadIdx.x & 31;
  int wave = threadIdx.x >> 5;
  long pair = (long)blockIdx.x * 8 + wave;
  if (pair >= (long)M * N_HEADS) return;
  int q = (int)(pair >> 3);
  int h = (int)(pair & 7);

  const float* vcol = value + h * HEAD_DIM + lane;
  float acc = 0.0f;

  for (int l = 0; l < N_LEVELS; ++l) {
    int H = shapes[2 * l + 0];
    int W = shapes[2 * l + 1];
    int start = starts[l];
    float rx = refp[((long)q * N_LEVELS + l) * 2 + 0];
    float ry = refp[((long)q * N_LEVELS + l) * 2 + 1];
    float fw = (float)W, fh = (float)H;

    for (int p = 0; p < N_POINTS; ++p) {
      long obase = (long)q * 256 + h * 32 + l * 8 + p * 2;
      float ox = offs[obase + 0];
      float oy = offs[obase + 1];
      float aw = attn[(long)q * 128 + h * 16 + l * 4 + p];

      // pixel = loc * size - 0.5 (align_corners=False, zero padding)
      float px = (rx + ox / fw) * fw - 0.5f;
      float py = (ry + oy / fh) * fh - 0.5f;
      float x0f = floorf(px), y0f = floorf(py);
      int x0 = (int)x0f, y0 = (int)y0f;
      float fx = px - x0f, fy = py - y0f;

      float w00 = (1.0f - fx) * (1.0f - fy);
      float w10 = fx * (1.0f - fy);
      float w01 = (1.0f - fx) * fy;
      float w11 = fx * fy;

      auto corner = [&](int xi, int yi, float w) {
        if (xi >= 0 && xi < W && yi >= 0 && yi < H) {
          acc += aw * w * vcol[(long)(start + yi * W + xi) * D_MODEL];
        }
      };
      corner(x0,     y0,     w00);
      corner(x0 + 1, y0,     w10);
      corner(x0,     y0 + 1, w01);
      corner(x0 + 1, y0 + 1, w11);
    }
  }
  tmp[(long)q * D_MODEL + h * HEAD_DIM + lane] = acc;
}

// ---------------------------------------------------------------------------
extern "C" void kernel_launch(void* const* d_in, const int* in_sizes, int n_in,
                              void* d_out, int out_size, void* d_ws, size_t ws_size,
                              hipStream_t stream) {
  const float* query   = (const float*)d_in[0];
  const float* refp    = (const float*)d_in[1];
  const float* flatten = (const float*)d_in[2];
  const int*   shapes  = (const int*)d_in[3];
  const int*   starts  = (const int*)d_in[4];
  const float* w_val   = (const float*)d_in[5];
  const float* b_val   = (const float*)d_in[6];
  const float* w_off   = (const float*)d_in[7];
  const float* b_off   = (const float*)d_in[8];
  const float* w_attn  = (const float*)d_in[9];
  const float* b_attn  = (const float*)d_in[10];
  const float* w_out   = (const float*)d_in[11];
  const float* b_out   = (const float*)d_in[12];
  float* out = (float*)d_out;

  const int M = in_sizes[0] / D_MODEL;           // 19947
  const int mtiles = (M + 15) / 16;              // 1247

  // Workspace carve (256B aligned slices)
  size_t cur = 0;
  auto take = [&](size_t bytes) -> char* {
    size_t r = (cur + 255) & ~(size_t)255;
    cur = r + bytes;
    return (char*)d_ws + r;
  };
  _Float16* A1    = (_Float16*)take((size_t)mtiles * 4096 * sizeof(_Float16));
  _Float16* B1    = (_Float16*)take((size_t)16 * 4096 * sizeof(_Float16));
  float*    valb  = (float*)take((size_t)M * 256 * sizeof(float));
  float*    offsb = (float*)take((size_t)M * 256 * sizeof(float));
  float*    attnb = (float*)take((size_t)M * 128 * sizeof(float));
  float*    tmpb  = (float*)take((size_t)M * 256 * sizeof(float));
  (void)ws_size;

  const long a_total  = (long)mtiles * 4096;
  const long b_total256 = 16L * 4096;
  const long b_total128 = 8L * 4096;
  dim3 blk(256);
  dim3 gA((unsigned)((a_total + 255) / 256));
  dim3 gB256((unsigned)((b_total256 + 255) / 256));
  dim3 gB128((unsigned)((b_total128 + 255) / 256));
  dim3 gGemm256((unsigned)mtiles, 2);   // N=256 -> 16 ntiles / 8 waves
  dim3 gGemm128((unsigned)mtiles, 1);   // N=128 ->  8 ntiles / 8 waves

  // 1-3: value = input_flatten @ value_proj_w + b
  msda_pack_a_kernel<<<gA, blk, 0, stream>>>(flatten, A1, M, a_total);
  msda_pack_b_kernel<<<gB256, blk, 0, stream>>>(w_val, B1, 256, b_total256);
  msda_wmma_gemm_kernel<<<gGemm256, blk, 0, stream>>>(
      (const v16h*)A1, (const v16h*)B1, b_val, valb, M, 256);

  // 4-6: offs = query @ samp_off_w + b
  msda_pack_b_kernel<<<gB256, blk, 0, stream>>>(w_off, B1, 256, b_total256);
  msda_pack_a_kernel<<<gA, blk, 0, stream>>>(query, A1, M, a_total);
  msda_wmma_gemm_kernel<<<gGemm256, blk, 0, stream>>>(
      (const v16h*)A1, (const v16h*)B1, b_off, offsb, M, 256);

  // 7-8: attn logits = query @ attn_w_w + b  (A1 already holds query)
  msda_pack_b_kernel<<<gB128, blk, 0, stream>>>(w_attn, B1, 128, b_total128);
  msda_wmma_gemm_kernel<<<gGemm128, blk, 0, stream>>>(
      (const v16h*)A1, (const v16h*)B1, b_attn, attnb, M, 128);

  // 9: softmax over 16 per (q, head)
  long rows = (long)M * N_HEADS;
  msda_softmax16_kernel<<<(unsigned)((rows + 255) / 256), blk, 0, stream>>>(attnb, M);

  // 10: deformable bilinear sampling -> tmp
  msda_sample_kernel<<<(unsigned)((rows + 7) / 8), blk, 0, stream>>>(
      valb, offsb, attnb, refp, shapes, starts, tmpb, M);

  // 11-13: out = tmp @ out_proj_w + b
  msda_pack_a_kernel<<<gA, blk, 0, stream>>>(tmpb, A1, M, a_total);
  msda_pack_b_kernel<<<gB256, blk, 0, stream>>>(w_out, B1, 256, b_total256);
  msda_wmma_gemm_kernel<<<gGemm256, blk, 0, stream>>>(
      (const v16h*)A1, (const v16h*)B1, b_out, out, M, 256);
}
